// SpatialCrossAttention_7052336300060
// MI455X (gfx1250) — compile-verified, matrix operand order checked by reading, not run on hardware
//
#include <hip/hip_runtime.h>
#include <hip/hip_bf16.h>

#define BS        2
#define NUM_CAMS  6
#define NUM_Q     2500
#define EMBED     256
#define NUM_K     4900
#define NUM_L     4

typedef __attribute__((ext_vector_type(16))) _Float16 v16h;
typedef __attribute__((ext_vector_type(8)))  _Float16 v8h;
typedef __attribute__((ext_vector_type(8)))  float    v8f;

#define QTILE   64      // queries per block (4 waves x 16)
#define KTILE   64      // keys per mainloop iteration
#define LDK     264     // halves per K row in LDS  (528 B, 16B aligned, bank-spread)
#define LDV     72      // halves per Vt row in LDS (144 B, 16B aligned, bank-spread)
#define LDS_S   72      // halves per P-scratch row

// LDS layout (in halves): Kt[64][LDK] | Vt[256][LDV] | Sh[4][16][LDS_S] | flag
#define K_HALVES   (64 * LDK)                 // 16896
#define V_HALVES   (256 * LDV)                // 18432
#define S_HALVES   (16 * LDS_S)               // 1152 per wave
#define SMEM_BYTES ((K_HALVES + V_HALVES + 4 * S_HALVES) * 2 + 16)  // 79888

// ---------------------------------------------------------------------------
// Kernel 1: derive per-camera validity (batch 0, as in reference) and counts
// ---------------------------------------------------------------------------
__global__ __launch_bounds__(256)
void mask_kernel(const int* __restrict__ bev, float* __restrict__ valid,
                 float* __restrict__ count)
{
    int q = blockIdx.x * blockDim.x + threadIdx.x;
    if (q >= NUM_Q) return;
    #pragma unroll
    for (int c = 0; c < NUM_CAMS; ++c) {
        const int* p = bev + (((size_t)c * BS + 0) * NUM_Q + q) * NUM_L;
        int s = p[0] + p[1] + p[2] + p[3];
        valid[c * NUM_Q + q] = (s > 0) ? 1.0f : 0.0f;
    }
    #pragma unroll
    for (int b = 0; b < BS; ++b) {
        float cnt = 0.0f;
        #pragma unroll
        for (int c = 0; c < NUM_CAMS; ++c) {
            const int* p = bev + (((size_t)c * BS + b) * NUM_Q + q) * NUM_L;
            int s = p[0] + p[1] + p[2] + p[3];
            cnt += (s > 0) ? 1.0f : 0.0f;
        }
        count[b * NUM_Q + q] = fmaxf(cnt, 1.0f);
    }
}

// ---------------------------------------------------------------------------
// Kernel 2: flash-attention per (b, cam, q-tile) with f16 WMMA, scatter-add
// into slots[b][q][d] for valid (cam,q) pairs.
// ---------------------------------------------------------------------------
__global__ __launch_bounds__(128)
void attn_kernel(const float* __restrict__ query, const float* __restrict__ key,
                 const float* __restrict__ value, const float* __restrict__ valid,
                 float* __restrict__ slots)
{
    extern __shared__ char smem_raw[];
    _Float16* Kt = (_Float16*)smem_raw;                 // [64][LDK]   keys (f16)
    _Float16* Vt = Kt + K_HALVES;                       // [256][LDV]  values transposed
    _Float16* Sh = Vt + V_HALVES;                       // 4 x [16][LDS_S] P scratch
    int* anyv = (int*)(smem_raw + (K_HALVES + V_HALVES + 4 * S_HALVES) * 2);

    const int tid  = threadIdx.x;
    const int wave = tid >> 5;
    const int lane = tid & 31;
    const int lm   = lane & 15;   // row/col within 16
    const int lg   = lane >> 4;   // half-wave group

    const int qt = blockIdx.x;
    const int bc = blockIdx.y;
    const int b  = bc / NUM_CAMS;
    const int c  = bc % NUM_CAMS;
    const int q0 = qt * QTILE;

    // ---- block-level early-out when no query in this tile is valid for cam c
    if (tid == 0) *anyv = 0;
    __syncthreads();
    if (tid < QTILE) {
        int q = q0 + tid;
        if (q < NUM_Q && valid[c * NUM_Q + q] > 0.5f) atomicOr(anyv, 1);
    }
    __syncthreads();
    if (*anyv == 0) return;

    // ---- load Q A-fragments (16 rows x 256 d), pre-scaled by 1/sqrt(256)
    const int qm = q0 + wave * 16 + lm;
    const int qc = (qm < NUM_Q) ? qm : (NUM_Q - 1);
    const float* qrow = query + ((size_t)b * NUM_Q + qc) * EMBED;
    const float scale = 0.0625f;

    v16h qa[8];
    #pragma unroll
    for (int f = 0; f < 8; ++f) {
        int d0 = f * 32 + lg * 8;
        #pragma unroll
        for (int e = 0; e < 8; ++e) {
            qa[f][e]     = (_Float16)(qrow[d0 + e]      * scale);
            qa[f][e + 8] = (_Float16)(qrow[d0 + 16 + e] * scale);
        }
    }

    v8f o[16] = {};            // 16 q-rows x 256 d accumulator (D layout)
    float mrow[8], lrow[8];
    #pragma unroll
    for (int r = 0; r < 8; ++r) { mrow[r] = -1e30f; lrow[r] = 0.0f; }

    const float* keyp = key   + (size_t)(c * BS + b) * NUM_K * EMBED;
    const float* valp = value + (size_t)(c * BS + b) * NUM_K * EMBED;
    _Float16* Shw = Sh + wave * S_HALVES;

    for (int kbase = 0; kbase < NUM_K; kbase += KTILE) {
        __syncthreads();   // previous tile fully consumed

        // ---- cooperative staging: K as [k][d] f16, V transposed as Vt[d][k]
        {
            int r  = tid >> 1;               // key row 0..63
            int cs = (tid & 1) * 128;        // d half 0 / 128
            int kg = kbase + r;
            _Float16* kdst = Kt + r * LDK + cs;
            if (kg < NUM_K) {
                const float* ksrc = keyp + (size_t)kg * EMBED + cs;
                #pragma unroll 4
                for (int i = 0; i < 128; i += 8) {
                    float4 a = *(const float4*)(ksrc + i);
                    float4 d = *(const float4*)(ksrc + i + 4);
                    v8h h;
                    h[0] = (_Float16)a.x; h[1] = (_Float16)a.y;
                    h[2] = (_Float16)a.z; h[3] = (_Float16)a.w;
                    h[4] = (_Float16)d.x; h[5] = (_Float16)d.y;
                    h[6] = (_Float16)d.z; h[7] = (_Float16)d.w;
                    *(v8h*)(kdst + i) = h;
                }
                const float* vsrc = valp + (size_t)kg * EMBED + cs;
                #pragma unroll 4
                for (int i = 0; i < 128; i += 4) {
                    float4 a = *(const float4*)(vsrc + i);
                    int d = cs + i;
                    Vt[(d + 0) * LDV + r] = (_Float16)a.x;
                    Vt[(d + 1) * LDV + r] = (_Float16)a.y;
                    Vt[(d + 2) * LDV + r] = (_Float16)a.z;
                    Vt[(d + 3) * LDV + r] = (_Float16)a.w;
                }
            } else {
                v8h z = {};
                #pragma unroll 4
                for (int i = 0; i < 128; i += 8) *(v8h*)(kdst + i) = z;
                #pragma unroll 4
                for (int i = 0; i < 128; ++i) Vt[(cs + i) * LDV + r] = (_Float16)0.0f;
            }
        }
        __syncthreads();

        // ---- S = Q K^T : 4 key-chunks of 16, 8 d-fragments each
        v8f sf[4];
        #pragma unroll
        for (int nc = 0; nc < 4; ++nc) {
            v8f acc = {};
            #pragma unroll
            for (int f = 0; f < 8; ++f) {
                const _Float16* kr = Kt + (nc * 16 + lm) * LDK + f * 32 + lg * 8;
                union { v16h v; v8h h[2]; } u;
                u.h[0] = *(const v8h*)(kr);
                u.h[1] = *(const v8h*)(kr + 16);
                acc = __builtin_amdgcn_wmma_f32_16x16x32_f16(
                        false, qa[f], false, u.v, (short)0, acc, false, false);
            }
            sf[nc] = acc;
        }

        // ---- mask keys beyond NUM_K (only last tile)
        #pragma unroll
        for (int nc = 0; nc < 4; ++nc) {
            if (kbase + nc * 16 + lm >= NUM_K) {
                #pragma unroll
                for (int r = 0; r < 8; ++r) sf[nc][r] = -1e30f;
            }
        }

        // ---- online softmax (rows live in 16-lane half-waves)
        float corr[8];
        #pragma unroll
        for (int r = 0; r < 8; ++r) {
            float mx = fmaxf(fmaxf(sf[0][r], sf[1][r]), fmaxf(sf[2][r], sf[3][r]));
            #pragma unroll
            for (int off = 8; off >= 1; off >>= 1)
                mx = fmaxf(mx, __shfl_xor(mx, off, 32));
            float mnew = fmaxf(mrow[r], mx);
            corr[r] = __expf(mrow[r] - mnew);
            mrow[r] = mnew;
            float rs = 0.0f;
            #pragma unroll
            for (int nc = 0; nc < 4; ++nc) {
                float p = __expf(sf[nc][r] - mnew);
                sf[nc][r] = p;
                rs += p;
            }
            #pragma unroll
            for (int off = 8; off >= 1; off >>= 1)
                rs += __shfl_xor(rs, off, 32);
            lrow[r] = lrow[r] * corr[r] + rs;
        }
        #pragma unroll
        for (int dc = 0; dc < 16; ++dc) {
            #pragma unroll
            for (int r = 0; r < 8; ++r) o[dc][r] *= corr[r];
        }

        // ---- D-layout -> A-layout via per-wave LDS scratch (same-wave DS in-order)
        #pragma unroll
        for (int nc = 0; nc < 4; ++nc) {
            #pragma unroll
            for (int r = 0; r < 8; ++r)
                Shw[(lg * 8 + r) * LDS_S + nc * 16 + lm] = (_Float16)sf[nc][r];
        }

        // ---- O += P V : 2 k-slices x 16 d-chunks
        #pragma unroll
        for (int ks = 0; ks < 2; ++ks) {
            const _Float16* pr = Shw + lm * LDS_S + ks * 32 + lg * 8;
            union { v16h v; v8h h[2]; } pu;
            pu.h[0] = *(const v8h*)(pr);
            pu.h[1] = *(const v8h*)(pr + 16);
            #pragma unroll
            for (int dc = 0; dc < 16; ++dc) {
                const _Float16* vr = Vt + (dc * 16 + lm) * LDV + ks * 32 + lg * 8;
                union { v16h v; v8h h[2]; } vu;
                vu.h[0] = *(const v8h*)(vr);
                vu.h[1] = *(const v8h*)(vr + 16);
                o[dc] = __builtin_amdgcn_wmma_f32_16x16x32_f16(
                          false, pu.v, false, vu.v, (short)0, o[dc], false, false);
            }
        }
    }

    // ---- epilogue: normalize by row sums, scatter-add valid rows
    float inv[8], vflag[8];
    #pragma unroll
    for (int r = 0; r < 8; ++r) {
        inv[r] = 1.0f / lrow[r];
        int qr = q0 + wave * 16 + lg * 8 + r;
        vflag[r] = (qr < NUM_Q) ? valid[c * NUM_Q + qr] : 0.0f;
    }
    #pragma unroll
    for (int dc = 0; dc < 16; ++dc) {
        int d = dc * 16 + lm;
        #pragma unroll
        for (int r = 0; r < 8; ++r) {
            if (vflag[r] > 0.5f) {
                int qr = q0 + wave * 16 + lg * 8 + r;
                atomicAdd(slots + ((size_t)b * NUM_Q + qr) * EMBED + d, o[dc][r] * inv[r]);
            }
        }
    }
}

// ---------------------------------------------------------------------------
// Kernel 3: out = (slots / count) @ W^T + b + query   (tiny GEMV per row)
// ---------------------------------------------------------------------------
__global__ __launch_bounds__(256)
void proj_kernel(const float* __restrict__ slots, const float* __restrict__ count,
                 const float* __restrict__ query, const float* __restrict__ W,
                 const float* __restrict__ bvec, float* __restrict__ out)
{
    __shared__ float s[EMBED];
    int bq  = blockIdx.x;           // b*NUM_Q + q
    int tid = threadIdx.x;
    float cnt = count[bq];
    s[tid] = slots[(size_t)bq * EMBED + tid] / cnt;
    __syncthreads();
    const float* wr = W + (size_t)tid * EMBED;
    float acc = bvec[tid];
    #pragma unroll 8
    for (int d = 0; d < EMBED; d += 4) {
        float4 w4 = *(const float4*)(wr + d);
        acc += w4.x * s[d] + w4.y * s[d + 1] + w4.z * s[d + 2] + w4.w * s[d + 3];
    }
    out[(size_t)bq * EMBED + tid] = acc + query[(size_t)bq * EMBED + tid];
}

// ---------------------------------------------------------------------------
extern "C" void kernel_launch(void* const* d_in, const int* in_sizes, int n_in,
                              void* d_out, int out_size, void* d_ws, size_t ws_size,
                              hipStream_t stream)
{
    const float* query = (const float*)d_in[0];
    const float* key   = (const float*)d_in[1];
    const float* value = (const float*)d_in[2];
    // d_in[3] = reference_points_cam (unused by reference math)
    const int*   bev   = (const int*)d_in[4];
    // d_in[5] = spatial_shapes (unused)
    const float* W     = (const float*)d_in[6];
    const float* bvec  = (const float*)d_in[7];
    float* out = (float*)d_out;

    char* ws = (char*)d_ws;
    float* valid = (float*)(ws);                  // 6*2500 floats
    float* count = (float*)(ws + 64 * 1024);      // 2*2500 floats
    float* slots = (float*)(ws + 128 * 1024);     // 2*2500*256 floats (~5 MB)

    hipMemsetAsync(slots, 0, (size_t)BS * NUM_Q * EMBED * sizeof(float), stream);

    mask_kernel<<<(NUM_Q + 255) / 256, 256, 0, stream>>>(bev, valid, count);

    dim3 grid((NUM_Q + QTILE - 1) / QTILE, BS * NUM_CAMS);
    attn_kernel<<<grid, 128, SMEM_BYTES, stream>>>(query, key, value, valid, slots);

    proj_kernel<<<BS * NUM_Q, 256, 0, stream>>>(slots, count, query, W, bvec, out);
}
